// PAdicRNNCell_12335146074822
// MI455X (gfx1250) — matrix operand order repaired and checked by reading, not compiled, not run on hardware
//
#include <hip/hip_runtime.h>
#include <hip/hip_bf16.h>

// ---------------------------------------------------------------------------
// PAdicRNNCell for MI455X (gfx1250): bf16 WMMA GEMMs + fused epilogues.
// B=4096, IN=1024, H=2560, P=10, D=256. ~230 GFLOP vs ~200MB traffic ->
// compute bound -> v_wmma_f32_16x16x32_bf16 for every matmul, async
// global->LDS DMA (ASYNCcnt) for double-buffered tile staging.
// ---------------------------------------------------------------------------

typedef __bf16 bf16_t;
typedef __attribute__((ext_vector_type(16))) __bf16 v16bf;
typedef __attribute__((ext_vector_type(8)))  __bf16 v8bf;
typedef __attribute__((ext_vector_type(8)))  float  v8f;
typedef __attribute__((ext_vector_type(4)))  int    v4i;

#define B_DIM  4096
#define IN_DIM 1024
#define H_DIM  2560
#define P_DIM  10
#define D_DIM  256

#if __has_builtin(__builtin_amdgcn_global_load_async_to_lds_b128)
#define USE_ASYNC_LDS 1
#endif

// 16-byte global -> LDS copy; async DMA (ASYNCcnt) when available.
// Builtin signature (from hipcc diag): (v4i AS1*, v4i AS3*, imm int, imm int).
__device__ __forceinline__ void copy16_g2l(const bf16_t* g, bf16_t* l) {
#ifdef USE_ASYNC_LDS
  typedef __attribute__((address_space(1))) v4i* gptr_t;
  typedef __attribute__((address_space(3))) v4i* lptr_t;
  // integer detour: C++ forbids direct cross-addrspace pointer casts; low 32
  // bits of a flat shared pointer are the AS3 offset on AMDGPU.
  __builtin_amdgcn_global_load_async_to_lds_b128(
      (gptr_t)(uintptr_t)g, (lptr_t)(unsigned)(uintptr_t)l, 0, 0);
#else
  *(v8bf*)l = *(const v8bf*)g;
#endif
}

__device__ __forceinline__ void async_wait0() {
#ifdef USE_ASYNC_LDS
#if __has_builtin(__builtin_amdgcn_s_wait_asynccnt)
  __builtin_amdgcn_s_wait_asynccnt(0);
#else
  asm volatile("s_wait_asynccnt 0x0" ::: "memory");
#endif
#endif
}

// ---------------------------------------------------------------------------
// f32 -> bf16 conversion (grid-stride)
// ---------------------------------------------------------------------------
__global__ void conv_f32_bf16(const float* __restrict__ src,
                              bf16_t* __restrict__ dst, int n) {
  int i = blockIdx.x * 256 + threadIdx.x;
  int stride = gridDim.x * 256;
  for (; i < n; i += stride) dst[i] = (bf16_t)src[i];
}

// ---------------------------------------------------------------------------
// WMMA GEMM: C(M,NN) = A(M,KK) @ W(NN,KK)^T  (bf16 in, f32 accum)
//  MODE 0: C = acc + bias[n]                       (f32 store)
//  MODE 1: C = sigmoid(C + acc)                    (f32 rmw)
//  MODE 2: C = tanh(C + acc)                       (f32 rmw)
//  MODE 3: p-adic: digit p = bn/2, A cols shifted by p*256, W = Wd[p];
//          Cb = bf16( (acc + bd[n]) * va[p] * R[m,n] )
// Block tile 128x128, K-step 32; 8 waves (2x4), each wave 64x32 (4x2 tiles).
// Double-buffered LDS, async global->LDS prefetch overlapped with WMMA.
// ---------------------------------------------------------------------------
template <int MODE, int NN, int KK, int LDA>
__global__ __launch_bounds__(256)
void gemm_wmma(const bf16_t* __restrict__ A, const bf16_t* __restrict__ Wt,
               const float* __restrict__ bias, float* __restrict__ C,
               bf16_t* __restrict__ Cb, const float* __restrict__ Rg,
               const float* __restrict__ va) {
  constexpr int LDSW = 40;  // 32 K-halves + 8 pad (80B rows, 16B aligned)
  constexpr int TILE = 128 * LDSW;
  __shared__ __align__(16) bf16_t As[2 * TILE];
  __shared__ __align__(16) bf16_t Ws[2 * TILE];

  const int tid  = threadIdx.x;
  const int lane = tid & 31;
  const int wave = tid >> 5;
  const int wm   = wave >> 2;  // 0..1  (M direction, 64 rows each)
  const int wn   = wave & 3;   // 0..3  (N direction, 32 cols each)
  const int bm   = blockIdx.y;
  const int bn   = blockIdx.x;

  const bf16_t* Wp = Wt;
  int a_col0 = 0;
  int w_row0 = bn * 128;
  int p      = 0;
  if (MODE == 3) {
    p      = bn >> 1;                          // digit index (128*2 == D)
    Wp     = Wt + (size_t)p * D_DIM * D_DIM;   // Wd[p] : (D,D) row-major
    a_col0 = p * D_DIM;                        // h digit slice
    w_row0 = (bn & 1) * 128;                   // local row inside Wd[p]
  }

  // staging coords: 512 chunks of 8 bf16 per tile, 2 per thread
  const int srow0 = tid >> 2;                 // rows 0..63
  const int sc8   = (tid & 3) * 8;
  const bf16_t* gA = A + (size_t)(bm * 128) * LDA + a_col0 + sc8;
  const bf16_t* gW = Wp + (size_t)w_row0 * KK + sc8;

  auto stage = [&](int buf, int k0) {
    bf16_t* as = As + buf * TILE;
    bf16_t* ws = Ws + buf * TILE;
#pragma unroll
    for (int j = 0; j < 2; ++j) {
      int row = srow0 + j * 64;
      copy16_g2l(gA + (size_t)row * LDA + k0, as + row * LDSW + sc8);
      copy16_g2l(gW + (size_t)row * KK + k0, ws + row * LDSW + sc8);
    }
  };

  const v8f vzero = {};
  v8f acc[4][2];
#pragma unroll
  for (int i = 0; i < 4; ++i)
#pragma unroll
    for (int j = 0; j < 2; ++j) acc[i][j] = vzero;

  const int frow = lane & 15;
  const int ahk  = (lane >> 4) * 8;   // A: K pair-halves 0..7 / 8..15
  const int bkb  = (lane >> 4) * 8;   // B: K pair-halves 0..7 / 8..15

  constexpr int NK = KK / 32;
  stage(0, 0);
  async_wait0();
  __syncthreads();

  for (int k = 0; k < NK; ++k) {
    const int cur = k & 1;
    if (k + 1 < NK) stage((k + 1) & 1, (k + 1) * 32);  // async prefetch

    const bf16_t* as = As + cur * TILE;
    const bf16_t* ws = Ws + cur * TILE;

    // A fragments: lane half selects K {0..7,16..23} vs {8..15,24..31}
    v16bf afrag[4];
#pragma unroll
    for (int i = 0; i < 4; ++i) {
      int r = wm * 64 + i * 16 + frow;
      ((v8bf*)&afrag[i])[0] = *(const v8bf*)&as[r * LDSW + ahk];
      ((v8bf*)&afrag[i])[1] = *(const v8bf*)&as[r * LDSW + 16 + ahk];
    }
    // B fragments: lane half selects K {0..7,16..23} vs {8..15,24..31}
    v16bf bfrag[2];
#pragma unroll
    for (int j = 0; j < 2; ++j) {
      int r = wn * 32 + j * 16 + frow;
      ((v8bf*)&bfrag[j])[0] = *(const v8bf*)&ws[r * LDSW + bkb];
      ((v8bf*)&bfrag[j])[1] = *(const v8bf*)&ws[r * LDSW + bkb + 16];
    }

#pragma unroll
    for (int i = 0; i < 4; ++i)
#pragma unroll
      for (int j = 0; j < 2; ++j)
        acc[i][j] = __builtin_amdgcn_wmma_f32_16x16x32_bf16(
            false, afrag[i], false, bfrag[j], (short)0, acc[i][j], false,
            false);

    async_wait0();   // prefetch for k+1 landed
    __syncthreads(); // everyone done reading buf cur before it is re-staged
  }

  // epilogue: C element r of v8f at lane l -> M = r + 8*(l>>4), N = l&15
  const int mo = (lane >> 4) * 8;
#pragma unroll
  for (int i = 0; i < 4; ++i) {
#pragma unroll
    for (int j = 0; j < 2; ++j) {
      const int n = bn * 128 + wn * 32 + j * 16 + frow;
      const int m0 = bm * 128 + wm * 64 + i * 16 + mo;
      const size_t base = (size_t)m0 * NN + n;
#pragma unroll
      for (int r = 0; r < 8; ++r) {
        const size_t idx = base + (size_t)r * NN;  // NN constexpr -> imm offs
        float v = acc[i][j][r];
        if (MODE == 0) {
          C[idx] = v + bias[n];
        } else if (MODE == 1) {
          float t = C[idx] + v;
          C[idx] = 1.0f / (1.0f + __expf(-t));
        } else if (MODE == 2) {
          float t = C[idx] + v;
          C[idx] = tanhf(t);
        } else {
          float t = (v + bias[n]) * va[p] * Rg[idx];
          Cb[idx] = (bf16_t)t;
        }
      }
    }
  }
}

// ---------------------------------------------------------------------------
// Final: h_new = z*h + (1-z)*htilde, then contractive row-norm projection.
// One 256-thread block per row (H=2560 -> 10 elems/thread).
// ---------------------------------------------------------------------------
__global__ __launch_bounds__(256)
void final_combine(const float* __restrict__ h, const float* __restrict__ z,
                   const float* __restrict__ ht, float* __restrict__ out) {
  __shared__ float red[256];
  const int b   = blockIdx.x;
  const int tid = threadIdx.x;
  float hn[10];
  float ss = 0.0f;
#pragma unroll
  for (int t = 0; t < 10; ++t) {
    size_t idx = (size_t)b * H_DIM + tid + t * 256;
    float zz = z[idx];
    float v  = zz * h[idx] + (1.0f - zz) * ht[idx];
    hn[t] = v;
    ss += v * v;
  }
  red[tid] = ss;
  __syncthreads();
  for (int s = 128; s > 0; s >>= 1) {
    if (tid < s) red[tid] += red[tid + s];
    __syncthreads();
  }
  const float norm  = sqrtf(red[0]);
  const float MAXN  = 100.0f;  // 1/(1-0.99)
  const float scale = (norm > MAXN) ? (MAXN / (norm + 1e-8f)) : 1.0f;
#pragma unroll
  for (int t = 0; t < 10; ++t)
    out[(size_t)b * H_DIM + tid + t * 256] = hn[t] * scale;
}

// ---------------------------------------------------------------------------
// Host launcher
// ---------------------------------------------------------------------------
extern "C" void kernel_launch(void* const* d_in, const int* in_sizes, int n_in,
                              void* d_out, int out_size, void* d_ws,
                              size_t ws_size, hipStream_t stream) {
  const float* x  = (const float*)d_in[0];
  const float* h  = (const float*)d_in[1];
  const float* Wz = (const float*)d_in[2];
  const float* bz = (const float*)d_in[3];
  const float* Uz = (const float*)d_in[4];
  const float* Wr = (const float*)d_in[5];
  const float* br = (const float*)d_in[6];
  const float* Ur = (const float*)d_in[7];
  const float* Wh = (const float*)d_in[8];
  const float* bh = (const float*)d_in[9];
  const float* Uh = (const float*)d_in[10];
  const float* Wd = (const float*)d_in[11];
  const float* bd = (const float*)d_in[12];
  const float* va = (const float*)d_in[13];
  float* out = (float*)d_out;

  char* ws = (char*)d_ws;
  auto alloc = [&](size_t bytes) -> void* {
    void* p = (void*)ws;
    ws += (bytes + 255) & ~(size_t)255;
    return p;
  };

  const size_t nX  = (size_t)B_DIM * IN_DIM;
  const size_t nH  = (size_t)B_DIM * H_DIM;
  const size_t nW  = (size_t)H_DIM * IN_DIM;
  const size_t nU  = (size_t)H_DIM * H_DIM;
  const size_t nWd = (size_t)P_DIM * D_DIM * D_DIM;

  bf16_t* xb  = (bf16_t*)alloc(nX * 2);
  bf16_t* hb  = (bf16_t*)alloc(nH * 2);
  bf16_t* Wzb = (bf16_t*)alloc(nW * 2);
  bf16_t* Wrb = (bf16_t*)alloc(nW * 2);
  bf16_t* Whb = (bf16_t*)alloc(nW * 2);
  bf16_t* Uzb = (bf16_t*)alloc(nU * 2);
  bf16_t* Urb = (bf16_t*)alloc(nU * 2);
  bf16_t* Uhb = (bf16_t*)alloc(nU * 2);
  bf16_t* Wdb = (bf16_t*)alloc(nWd * 2);
  float*  zpre = (float*)alloc(nH * 4);   // -> z
  float*  rpre = (float*)alloc(nH * 4);   // -> r
  float*  hx   = (float*)alloc(nH * 4);   // x@Wh.T+bh -> htilde
  bf16_t* rpb  = (bf16_t*)alloc(nH * 2);  // (r * h_padic) in bf16

  auto conv = [&](const float* s, bf16_t* d, size_t n) {
    int blocks = (int)((n + 1023) / 1024);
    conv_f32_bf16<<<blocks, 256, 0, stream>>>(s, d, (int)n);
  };
  conv(x, xb, nX);
  conv(h, hb, nH);
  conv(Wz, Wzb, nW);
  conv(Wr, Wrb, nW);
  conv(Wh, Whb, nW);
  conv(Uz, Uzb, nU);
  conv(Ur, Urb, nU);
  conv(Uh, Uhb, nU);
  conv(Wd, Wdb, nWd);

  const dim3 blk(256);
  const dim3 grd(H_DIM / 128, B_DIM / 128);  // (N/128, M/128)

  // x-side GEMMs with bias
  gemm_wmma<0, H_DIM, IN_DIM, IN_DIM><<<grd, blk, 0, stream>>>(
      xb, Whb, bh, hx, nullptr, nullptr, nullptr);
  gemm_wmma<0, H_DIM, IN_DIM, IN_DIM><<<grd, blk, 0, stream>>>(
      xb, Wzb, bz, zpre, nullptr, nullptr, nullptr);
  gemm_wmma<0, H_DIM, IN_DIM, IN_DIM><<<grd, blk, 0, stream>>>(
      xb, Wrb, br, rpre, nullptr, nullptr, nullptr);

  // gates: z = sigmoid(zpre + h@Uz.T), r = sigmoid(rpre + h@Ur.T)
  gemm_wmma<1, H_DIM, H_DIM, H_DIM><<<grd, blk, 0, stream>>>(
      hb, Uzb, nullptr, zpre, nullptr, nullptr, nullptr);
  gemm_wmma<1, H_DIM, H_DIM, H_DIM><<<grd, blk, 0, stream>>>(
      hb, Urb, nullptr, rpre, nullptr, nullptr, nullptr);

  // p-adic digit GEMMs: rpb = bf16(r * va[p] * (h_digit @ Wd[p].T + bd[p]))
  gemm_wmma<3, H_DIM, D_DIM, H_DIM><<<grd, blk, 0, stream>>>(
      hb, Wdb, bd, nullptr, rpb, rpre, va);

  // htilde = tanh(hx + rpb @ Uh.T)
  gemm_wmma<2, H_DIM, H_DIM, H_DIM><<<grd, blk, 0, stream>>>(
      rpb, Uhb, nullptr, hx, nullptr, nullptr, nullptr);

  // combine + contractive projection
  final_combine<<<B_DIM, 256, 0, stream>>>(h, zpre, hx, out);
}
